// FastSlidingWindowModel_3693671874778
// MI455X (gfx1250) — compile-verified, matrix operand order checked by reading, not run on hardware
//
#include <hip/hip_runtime.h>
#include <hip/hip_bf16.h>
#include <stdint.h>

// ---------------------------------------------------------------------------
// Sliding-window attention, MI455X (gfx1250), wave32 + WMMA bf16.
//   D_MODEL=1024, NHEAD=16, HEAD_DIM=64, WINDOW=256, B=2, S=2048
// ---------------------------------------------------------------------------

#define D_MODEL 1024
#define NHEAD   16
#define HEAD_DIM 64
#define WINDOW  256
#define BATCH   2
#define SEQ     2048
#define NTOK    (BATCH * SEQ)   // 4096

typedef __bf16 bf16;
typedef __attribute__((ext_vector_type(16))) __bf16 v16bf;
typedef __attribute__((ext_vector_type(8)))  __bf16 v8bf;
typedef __attribute__((ext_vector_type(8)))  float  v8f;
typedef __attribute__((ext_vector_type(4)))  float  v4f;
typedef __attribute__((ext_vector_type(2)))  int    v2i;

// --- gfx1250 async global->LDS path (guarded; sync fallback keeps same CFG) -
#ifndef USE_ASYNC_LDS
#if __has_builtin(__builtin_amdgcn_global_load_async_to_lds_b64)
#define USE_ASYNC_LDS 1
#else
#define USE_ASYNC_LDS 0
#endif
#endif

#if USE_ASYNC_LDS
#if __has_builtin(__builtin_amdgcn_s_wait_asynccnt)
#define WAIT_ASYNC() __builtin_amdgcn_s_wait_asynccnt(0)
#else
#define WAIT_ASYNC() asm volatile("s_wait_asynccnt 0" ::: "memory")
#endif
static __device__ inline void copy8_g2l(const bf16* gsrc, bf16* ldst) {
    // builtin takes pointers to 8-byte v2i payloads (per compiler diagnostic)
    __builtin_amdgcn_global_load_async_to_lds_b64(
        (v2i*)(void*)gsrc, (v2i*)(void*)ldst, 0, 0);
}
#else
#define WAIT_ASYNC() ((void)0)
static __device__ inline void copy8_g2l(const bf16* gsrc, bf16* ldst) {
    *(uint64_t*)ldst = *(const uint64_t*)gsrc;   // sync fallback
}
#endif

// Build a 16-bit A/B WMMA fragment from two 16-byte row chunks.
// Element e of v16bf holds K = kb8 + (e<8 ? e : e+8) per the ISA 16-bit
// 16x32 A-matrix layout (lanes 0-15: kb8=0, lanes 16-31: kb8=8).
static __device__ inline v16bf make_frag(const bf16* p_lo, const bf16* p_hi) {
    v8bf lo = *(const v8bf*)p_lo;
    v8bf hi = *(const v8bf*)p_hi;
    v16bf r;
#pragma unroll
    for (int i = 0; i < 8; ++i) { r[i] = lo[i]; r[i + 8] = hi[i]; }
    return r;
}

static __device__ inline v8f wmma_bf16(v16bf a, v16bf b, v8f c) {
    return __builtin_amdgcn_wmma_f32_16x16x32_bf16(
        /*neg_a=*/false, a, /*neg_b=*/false, b,
        /*c_mod=*/(short)0, c, /*reuse_a=*/false, /*reuse_b=*/false);
}

// ---------------------------------------------------------------------------
// f32 -> bf16 conversion, 8 elements/thread (16-byte bf16 stores)
// ---------------------------------------------------------------------------
__global__ void cvt_f32_to_bf16(const float* __restrict__ in,
                                bf16* __restrict__ out, int n8) {
    for (int i = blockIdx.x * blockDim.x + threadIdx.x; i < n8;
         i += gridDim.x * blockDim.x) {
        v4f a = ((const v4f*)in)[2 * i];
        v4f b = ((const v4f*)in)[2 * i + 1];
        v8bf o;
#pragma unroll
        for (int j = 0; j < 4; ++j) { o[j] = (bf16)a[j]; o[4 + j] = (bf16)b[j]; }
        ((v8bf*)out)[i] = o;
    }
}

// ---------------------------------------------------------------------------
// WMMA GEMM: C[m,n] = sum_k A[m,k] * B[n,k]   (A:[M,K], B:[N,K] bf16)
// 32x64 tile per wave; 8 waves/block share one m-tile, so the 32x32 A k-chunk
// is staged in LDS (double-buffered, async global->LDS copy) and B fragments
// are ping-pong register buffered (k-loop unrolled by 2: no copy movs).
// Output layouts:
//   mode 0: f32  out[m*N + n]
//   mode 1: bf16 out[((b*H + h)*S + s)*64 + d] * scale   (Q/K, head-major)
//   mode 2: bf16 out[((b*H + h)*64 + d)*S + s]           (V transposed)
// Requires K % 64 == 0 (nK even).
// ---------------------------------------------------------------------------
__global__ __launch_bounds__(256)
void gemm_bf16_wmma(const bf16* __restrict__ A, const bf16* __restrict__ B,
                    void* __restrict__ out, int M, int N, int K,
                    int mode, float scale) {
    __shared__ bf16 ldsA[2][32 * 32];      // two 2 KB A-chunk buffers

    const int tid   = threadIdx.x;
    const int lane  = tid & 31;
    const int wave  = tid >> 5;
    const int gid   = blockIdx.x * 8 + wave;
    const int tilesN = N >> 6;
    const int m0 = (gid / tilesN) << 5;    // 32-row tile (same for all 8 waves)
    const int n0 = (gid % tilesN) << 6;

    const int lan  = lane & 15;
    const int half = lane >> 4;
    const int kb8  = half << 3;
    const int nK   = K >> 5;

    // cooperative A-copy coords: 256 threads x 8 bytes = 2048 B per chunk
    const int crow = tid >> 3;             // 0..31
    const int ccol = (tid & 7) << 2;       // 0,4,...,28 (bf16 elems)
    const bf16* gcopy = A + (size_t)(m0 + crow) * K + ccol;

    // prologue: stage chunk 0
    copy8_g2l(gcopy, &ldsA[0][crow * 32 + ccol]);

    // B row pointers + first B fragments into ping buffer
    const bf16* brow[4];
#pragma unroll
    for (int t = 0; t < 4; ++t)
        brow[t] = B + (size_t)(n0 + (t << 4) + lan) * K;
    v16bf bA[4], bB[4];
#pragma unroll
    for (int t = 0; t < 4; ++t)
        bA[t] = make_frag(brow[t] + kb8, brow[t] + kb8 + 16);

    v8f acc[2][4];
#pragma unroll
    for (int f = 0; f < 2; ++f)
#pragma unroll
        for (int t = 0; t < 4; ++t)
            acc[f][t] = (v8f){0.f,0.f,0.f,0.f,0.f,0.f,0.f,0.f};

    // stage A chunk kc_next + load B fragments for kc_next into bdst
    auto stage = [&](int kc_next, v16bf (&bdst)[4]) {
        copy8_g2l(gcopy + (kc_next << 5),
                  &ldsA[kc_next & 1][crow * 32 + ccol]);
        const int ko = kc_next << 5;
        __builtin_prefetch(brow[0] + ko + 32, 0, 1);
#pragma unroll
        for (int t = 0; t < 4; ++t)
            bdst[t] = make_frag(brow[t] + ko + kb8, brow[t] + ko + kb8 + 16);
    };
    // consume LDS A chunk kc with B fragments bsrc
    auto compute = [&](int kc, v16bf (&bsrc)[4]) {
        const bf16* ar0 = &ldsA[kc & 1][lan * 32];
        const bf16* ar1 = &ldsA[kc & 1][(16 + lan) * 32];
        v16bf a0 = make_frag(ar0 + kb8, ar0 + kb8 + 16);
        v16bf a1 = make_frag(ar1 + kb8, ar1 + kb8 + 16);
#pragma unroll
        for (int t = 0; t < 4; ++t) {
            acc[0][t] = wmma_bf16(a0, bsrc[t], acc[0][t]);
            acc[1][t] = wmma_bf16(a1, bsrc[t], acc[1][t]);
        }
    };

    for (int kc = 0; kc < nK; kc += 2) {
        // even chunk: consume bA, stage into bB
        WAIT_ASYNC();            // my chunk-kc copy landed
        __syncthreads();         // everyone's copy landed / prior reads done
        if (kc + 1 < nK) stage(kc + 1, bB);
        compute(kc, bA);

        // odd chunk: consume bB, stage into bA
        WAIT_ASYNC();
        __syncthreads();
        if (kc + 2 < nK) stage(kc + 2, bA);
        compute(kc + 1, bB);
    }

    // C-layout: acc[f][t] elem r -> row m0 + 16f + r + 8*half, col n0+16t+lan
#pragma unroll
    for (int f = 0; f < 2; ++f)
#pragma unroll
        for (int t = 0; t < 4; ++t)
#pragma unroll
            for (int r = 0; r < 8; ++r) {
                const int row = m0 + (f << 4) + r + (half << 3);
                const int col = n0 + (t << 4) + lan;
                const float v = acc[f][t][r] * scale;
                if (mode == 0) {
                    ((float*)out)[(size_t)row * N + col] = v;
                } else if (mode == 1) {
                    const int b = row >> 11, s = row & (SEQ - 1);
                    const int h = col >> 6, d = col & (HEAD_DIM - 1);
                    ((bf16*)out)[((size_t)(b * NHEAD + h) * SEQ + s) * HEAD_DIM + d] = (bf16)v;
                } else {
                    const int b = row >> 11, s = row & (SEQ - 1);
                    const int h = col >> 6, d = col & (HEAD_DIM - 1);
                    ((bf16*)out)[((size_t)(b * NHEAD + h) * HEAD_DIM + d) * SEQ + s] = (bf16)v;
                }
            }
}

// ---------------------------------------------------------------------------
// Fused sliding-window attention (flash-style online softmax).
// One wave per (b, h, 16-query tile).  Q pre-scaled by 1/sqrt(HEAD_DIM).
//   Qh,Kh: [B,H,S,64] bf16     Vt: [B,H,64,S] bf16
//   out  : bf16 [B*S, D_MODEL] (attention output, pre output-projection)
// ---------------------------------------------------------------------------
#define LDS_STRIDE 40  // bf16 elems per row (80B, 16B aligned, bank-spread)

__global__ __launch_bounds__(256)
void attn_swa_wmma(const bf16* __restrict__ Qh, const bf16* __restrict__ Kh,
                   const bf16* __restrict__ Vt, bf16* __restrict__ out) {
    __shared__ bf16 lds_p[8 * 16 * LDS_STRIDE];

    const int lane = threadIdx.x & 31;
    const int wave = threadIdx.x >> 5;
    const int gid  = blockIdx.x * 8 + wave;

    const int qt = gid & 127;            // S/16 = 128 query tiles
    const int h  = (gid >> 7) & 15;
    const int b  = gid >> 11;
    const int q0 = qt << 4;

    const int lan  = lane & 15;
    const int half = lane >> 4;
    const int kb8  = half << 3;

    const bf16* Qhead = Qh + ((size_t)(b * NHEAD + h) * SEQ) * HEAD_DIM;
    const bf16* Khead = Kh + ((size_t)(b * NHEAD + h) * SEQ) * HEAD_DIM;
    const bf16* Vhead = Vt + ((size_t)(b * NHEAD + h) * HEAD_DIM) * SEQ;
    bf16* ldsw = lds_p + wave * 16 * LDS_STRIDE;

    // Q A-fragments for the two head-dim chunks (lane row = q0 + lan)
    const bf16* qrow = Qhead + (size_t)(q0 + lan) * HEAD_DIM;
    v16bf a_q[2];
#pragma unroll
    for (int c = 0; c < 2; ++c)
        a_q[c] = make_frag(qrow + (c << 5) + kb8, qrow + (c << 5) + kb8 + 16);

    v8f acc[4];
#pragma unroll
    for (int t = 0; t < 4; ++t) acc[t] = (v8f){0.f,0.f,0.f,0.f,0.f,0.f,0.f,0.f};
    float rmax[8], rsum[8];
#pragma unroll
    for (int r = 0; r < 8; ++r) { rmax[r] = -1e30f; rsum[r] = 0.f; }

    // 9 key-chunks of 32 cover keys [q0-256, q0+31] ⊇ band of all 16 queries
    for (int kc = 0; kc < 9; ++kc) {
        const int kb = q0 - WINDOW + (kc << 5);

        // --- scores: 2 key tiles x 2 head-dim chunks -----------------------
        v8f s0 = (v8f){0.f,0.f,0.f,0.f,0.f,0.f,0.f,0.f};
        v8f s1 = s0;
#pragma unroll
        for (int c = 0; c < 2; ++c) {
            const int k0 = kb + lan;
            const int k1 = kb + 16 + lan;
            const int k0c = min(max(k0, 0), SEQ - 1);
            const int k1c = min(max(k1, 0), SEQ - 1);
            const bf16* kr0 = Khead + (size_t)k0c * HEAD_DIM + (c << 5);
            const bf16* kr1 = Khead + (size_t)k1c * HEAD_DIM + (c << 5);
            v16bf b0 = make_frag(kr0 + kb8, kr0 + kb8 + 16);
            v16bf b1 = make_frag(kr1 + kb8, kr1 + kb8 + 16);
            s0 = wmma_bf16(a_q[c], b0, s0);
            s1 = wmma_bf16(a_q[c], b1, s1);
        }

        // --- band mask + chunk row-max -------------------------------------
        const int c0 = kb + lan;
        const int c1 = kb + 16 + lan;
        float sm0[8], sm1[8], cmax[8];
        bool  v0[8], v1[8];
#pragma unroll
        for (int r = 0; r < 8; ++r) {
            const int row = q0 + r + (half << 3);
            const int d0 = row - c0, d1 = row - c1;
            v0[r] = (c0 >= 0) & (d0 >= 0) & (d0 < WINDOW);
            v1[r] = (c1 >= 0) & (d1 >= 0) & (d1 < WINDOW);
            sm0[r] = v0[r] ? s0[r] : -1e30f;
            sm1[r] = v1[r] ? s1[r] : -1e30f;
            cmax[r] = fmaxf(sm0[r], sm1[r]);
        }
        // reduce max across the 16 lanes of this half-wave
#pragma unroll
        for (int m = 1; m < 16; m <<= 1)
#pragma unroll
            for (int r = 0; r < 8; ++r)
                cmax[r] = fmaxf(cmax[r], __shfl_xor(cmax[r], m, 32));

        // --- online softmax rescale ---------------------------------------
        float fac[8];
#pragma unroll
        for (int r = 0; r < 8; ++r) {
            const float mnew = fmaxf(rmax[r], cmax[r]);
            fac[r]  = __expf(rmax[r] - mnew);
            rmax[r] = mnew;
        }
#pragma unroll
        for (int t = 0; t < 4; ++t)
#pragma unroll
            for (int r = 0; r < 8; ++r) acc[t][r] *= fac[r];

        // --- probabilities -> LDS (C-layout -> row-major 16x32 bf16) -------
#pragma unroll
        for (int r = 0; r < 8; ++r) {
            const float p0 = v0[r] ? __expf(sm0[r] - rmax[r]) : 0.f;
            const float p1 = v1[r] ? __expf(sm1[r] - rmax[r]) : 0.f;
            rsum[r] = rsum[r] * fac[r] + p0 + p1;
            const int row = r + (half << 3);
            ldsw[row * LDS_STRIDE + lan]      = (bf16)p0;
            ldsw[row * LDS_STRIDE + 16 + lan] = (bf16)p1;
        }
        asm volatile("s_wait_dscnt 0" ::: "memory");  // per-wave DS ordering

        // --- read back as 16x32 A-fragment (lane row = lan) ---------------
        const bf16* prow = ldsw + lan * LDS_STRIDE;
        v16bf a_p = make_frag(prow + kb8, prow + kb8 + 16);

        // --- PV: 4 output tiles of 16 head-dims ---------------------------
#pragma unroll
        for (int t = 0; t < 4; ++t) {
            const int d = (t << 4) + lan;                   // B col = head dim
            const bf16* vrow = Vhead + (size_t)d * SEQ;
            const int s0i = min(max(kb + kb8, 0), SEQ - 8);
            const int s1i = min(max(kb + kb8 + 16, 0), SEQ - 8);
            v16bf bv = make_frag(vrow + s0i, vrow + s1i);
            acc[t] = wmma_bf16(a_p, bv, acc[t]);
        }
    }

    // --- finalize: reduce row sums across half-wave, normalize, store -----
#pragma unroll
    for (int m = 1; m < 16; m <<= 1)
#pragma unroll
        for (int r = 0; r < 8; ++r)
            rsum[r] += __shfl_xor(rsum[r], m, 32);
    float inv[8];
#pragma unroll
    for (int r = 0; r < 8; ++r) inv[r] = 1.0f / rsum[r];

#pragma unroll
    for (int t = 0; t < 4; ++t)
#pragma unroll
        for (int r = 0; r < 8; ++r) {
            const int row = q0 + r + (half << 3);
            const int col = h * HEAD_DIM + (t << 4) + lan;
            out[((size_t)b * SEQ + row) * D_MODEL + col] = (bf16)(acc[t][r] * inv[r]);
        }
}

// ---------------------------------------------------------------------------
// Host launcher
// ---------------------------------------------------------------------------
extern "C" void kernel_launch(void* const* d_in, const int* in_sizes, int n_in,
                              void* d_out, int out_size, void* d_ws, size_t ws_size,
                              hipStream_t stream) {
    const float* x  = (const float*)d_in[0];
    const float* Wq = (const float*)d_in[1];
    const float* Wk = (const float*)d_in[2];
    const float* Wv = (const float*)d_in[3];
    const float* Wo = (const float*)d_in[4];

    char* ws = (char*)d_ws;
    const size_t MB = 1024 * 1024;
    bf16* xb    = (bf16*)(ws + 0);        //  8 MB  [4096,1024]
    bf16* wqb   = (bf16*)(ws + 8 * MB);   //  2 MB
    bf16* wkb   = (bf16*)(ws + 10 * MB);  //  2 MB
    bf16* wvb   = (bf16*)(ws + 12 * MB);  //  2 MB
    bf16* wob   = (bf16*)(ws + 14 * MB);  //  2 MB
    bf16* qb    = (bf16*)(ws + 16 * MB);  //  8 MB  [B,H,S,64] (pre-scaled)
    bf16* kb    = (bf16*)(ws + 24 * MB);  //  8 MB  [B,H,S,64]
    bf16* vtb   = (bf16*)(ws + 32 * MB);  //  8 MB  [B,H,64,S]
    bf16* attnb = (bf16*)(ws + 40 * MB);  //  8 MB  [4096,1024]

    const int nx8 = (NTOK * D_MODEL) / 8;     // 512K vec8 chunks
    const int nw8 = (D_MODEL * D_MODEL) / 8;  // 128K vec8 chunks
    cvt_f32_to_bf16<<<1024, 256, 0, stream>>>(x,  xb,  nx8);
    cvt_f32_to_bf16<<<512,  256, 0, stream>>>(Wq, wqb, nw8);
    cvt_f32_to_bf16<<<512,  256, 0, stream>>>(Wk, wkb, nw8);
    cvt_f32_to_bf16<<<512,  256, 0, stream>>>(Wv, wvb, nw8);
    cvt_f32_to_bf16<<<512,  256, 0, stream>>>(Wo, wob, nw8);

    // (M/32)*(N/64) = 128*16 = 2048 waves -> 256 blocks of 8 waves
    const float qscale = 0.125f;  // 1/sqrt(HEAD_DIM)
    gemm_bf16_wmma<<<256, 256, 0, stream>>>(xb, wqb, qb,  NTOK, D_MODEL, D_MODEL, 1, qscale);
    gemm_bf16_wmma<<<256, 256, 0, stream>>>(xb, wkb, kb,  NTOK, D_MODEL, D_MODEL, 1, 1.0f);
    gemm_bf16_wmma<<<256, 256, 0, stream>>>(xb, wvb, vtb, NTOK, D_MODEL, D_MODEL, 2, 1.0f);

    // B*H*(S/16) = 4096 waves -> 512 blocks
    attn_swa_wmma<<<512, 256, 0, stream>>>(qb, kb, vtb, attnb);

    gemm_bf16_wmma<<<256, 256, 0, stream>>>(attnb, wob, d_out, NTOK, D_MODEL, D_MODEL, 0, 1.0f);
}
